// Refiner_88948772700682
// MI455X (gfx1250) — matrix operand order repaired and checked by reading, not compiled, not run on hardware
//
#include <hip/hip_runtime.h>
#include <hip/hip_bf16.h>
#include <math.h>

// ---------------------------------------------------------------------------
// Problem constants (match reference)
// ---------------------------------------------------------------------------
#define PB   8
#define PN   2048
#define PC   128
#define PK   16
#define PKMAX 32
#define PM   (PB * PN)          // 16384 rows
#define NEG_SLOPE 0.2f

typedef _Float16 v16h __attribute__((ext_vector_type(16)));
typedef _Float16 v8h  __attribute__((ext_vector_type(8)));
typedef _Float16 v4h  __attribute__((ext_vector_type(4)));
typedef float    v8f  __attribute__((ext_vector_type(8)));

constexpr int FLAG_LEAKY = 1;
constexpr int FLAG_ACCUM = 2;

// ---------------------------------------------------------------------------
// Kernel 1: brute-force KNN, one block (256 threads) per (b, n) row.
// ---------------------------------------------------------------------------
__global__ void knn_kernel(const float* __restrict__ pos, int* __restrict__ nbr)
{
    __shared__ float d2[PN];
    __shared__ float rv[256];
    __shared__ int   ri[256];

    const int bn = blockIdx.x;
    const int b  = bn / PN;
    const int n  = bn % PN;
    const int t  = threadIdx.x;

    const float* pb = pos + (size_t)b * PN * 3;
    const float pi0 = pb[n * 3 + 0];
    const float pi1 = pb[n * 3 + 1];
    const float pi2 = pb[n * 3 + 2];

    for (int j = t; j < PN; j += 256) {
        float dx = pi0 - pb[j * 3 + 0];
        float dy = pi1 - pb[j * 3 + 1];
        float dz = pi2 - pb[j * 3 + 2];
        float d  = dx * dx + dy * dy + dz * dz;
        if (j == n) d = 1.0e10f;
        d2[j] = d;
    }
    __syncthreads();

    for (int k = 0; k < PKMAX; ++k) {
        float bestv = 3.0e38f;
        int   besti = PN;
        for (int j = t; j < PN; j += 256) {
            float v = d2[j];
            if (v < bestv || (v == bestv && j < besti)) { bestv = v; besti = j; }
        }
        rv[t] = bestv; ri[t] = besti;
        __syncthreads();
        for (int s = 128; s > 0; s >>= 1) {
            if (t < s) {
                if (rv[t + s] < rv[t] || (rv[t + s] == rv[t] && ri[t + s] < ri[t])) {
                    rv[t] = rv[t + s]; ri[t] = ri[t + s];
                }
            }
            __syncthreads();
        }
        if (t == 0) {
            nbr[(size_t)bn * PKMAX + k] = ri[0];
            d2[ri[0]] = 3.0e38f;
        }
        __syncthreads();
    }
}

// ---------------------------------------------------------------------------
// Pack kernels: one-shot operand preparation so the GEMM streams b128.
// ---------------------------------------------------------------------------
// W [K x COUT] f32 row-major -> WMMA B fragments (lane-ordered v16h).
// B 32x16 f16 layout: lane n = lane&15; half i -> K = kk + (lane>>4)*8 + i (i<8), kk + 16 + (lane>>4)*8 + (i-8) (i>=8).
template<int K, int COUT>
__global__ void pack_w_kernel(const float* __restrict__ W, v16h* __restrict__ PW)
{
    constexpr int NCT = (COUT + 15) / 16;
    const int lane = threadIdx.x & 31;
    const int mloc = lane & 15;
    const int hi   = lane >> 4;
    const int kc   = blockIdx.x;                 // K chunk (0..ceil(K/32)-1)
    const int ct   = blockIdx.y;                 // column tile
    const int kk   = kc * 32;

    int ncol = ct * 16 + mloc;
    int ncl  = (COUT % 16 != 0) ? (ncol < COUT ? ncol : COUT - 1) : ncol;

    v16h bfrag;
#pragma unroll
    for (int i = 0; i < 16; ++i) {
        int kb  = kk + ((i < 8) ? (hi * 8 + i) : (16 + hi * 8 + (i - 8)));
        int kbc = kb < K ? kb : K - 1;
        float bv = W[(size_t)kbc * COUT + ncl];
        bfrag[i] = (_Float16)((kb < K) ? bv : 0.0f);
    }
    PW[((size_t)kc * NCT + ct) * 32 + lane] = bfrag;
}

// f32 -> f16 row-major activation conversion (4 elements / thread).
__global__ void pack_f16_kernel(const float* __restrict__ A,
                                _Float16* __restrict__ O, int n4)
{
    int i = blockIdx.x * blockDim.x + threadIdx.x;
    if (i < n4) {
        float4 q = ((const float4*)A)[i];
        v4h o;
        o[0] = (_Float16)q.x; o[1] = (_Float16)q.y;
        o[2] = (_Float16)q.z; o[3] = (_Float16)q.w;
        ((v4h*)O)[i] = o;
    }
}

// ---------------------------------------------------------------------------
// Kernel 2: WMMA GEMM on pre-packed operands.
//   Y[M x COUT] = epilogue( [A16 | A2][M x K] * W[K x COUT] )
// A16: f16 row-major [M x K1]; A2: f32 tail columns (K1..K-1), fusion only.
// PW : packed B fragments from pack_w_kernel (tail chunk pre-zero-padded).
// One wave computes a 16 x (16*NTPB) stripe; 4 waves/block on row slabs.
// Fragment layouts per CDNA5 ISA 7.12.2 (wave32):
//   A 16x32 f16 : lane m=lane&15, hi=lane>>4; two contiguous 8-half runs
//                 at kk+hi*8 and kk+16+hi*8  -> 2x global_load_b128
//   C/D 16x16 f32: VGPR r -> M = r + 8*hi, N = lane&15
// ---------------------------------------------------------------------------
template<int K, int K1, int A2W, int COUT, int YLD, int ADDLD, int NTPB,
         int FLAGS, bool HASBIAS, bool HASADD>
__global__ void gemm_wmma(const _Float16* __restrict__ A16,
                          const float* __restrict__ A2,
                          const v16h* __restrict__ PW,
                          const float* __restrict__ bias,
                          const float* __restrict__ addend,
                          float* __restrict__ Y)
{
    constexpr int NCT = (COUT + 15) / 16;

    const int lane   = threadIdx.x & 31;
    const int wv     = threadIdx.x >> 5;            // wave in block (0..3)
    const int mloc   = lane & 15;
    const int hi     = lane >> 4;
    const int rowblk = blockIdx.x * 4 + wv;         // 16-row slab index
    const int row    = rowblk * 16 + mloc;
    const int colt0  = blockIdx.y * NTPB;

    v8f acc[NTPB];
#pragma unroll
    for (int t = 0; t < NTPB; ++t)
#pragma unroll
        for (int r = 0; r < 8; ++r) acc[t][r] = 0.0f;

    const _Float16* arow = A16 + (size_t)row * K1;

    // ---- full K chunks of 32: 2x b128 A loads, 2x b128 B loads/tile ----
#pragma unroll
    for (int kc = 0; kc < K / 32; ++kc) {
        const int kk = kc * 32;
        v16h afrag;
        {
            const v8h lo = *(const v8h*)(arow + kk + hi * 8);
            const v8h hs = *(const v8h*)(arow + kk + 16 + hi * 8);
#pragma unroll
            for (int i = 0; i < 8; ++i) { afrag[i] = lo[i]; afrag[8 + i] = hs[i]; }
        }
#pragma unroll
        for (int t = 0; t < NTPB; ++t) {
            v16h bfrag = PW[((size_t)kc * NCT + colt0 + t) * 32 + lane];
            acc[t] = __builtin_amdgcn_wmma_f32_16x16x32_f16(
                         false, afrag, false, bfrag, (short)0, acc[t],
                         false, false);
        }
    }

    // ---- tail chunk (K % 32 != 0): A tail from f32 A2 (clamped + cndmask);
    //      B tail already zero-padded inside PW ----
    if constexpr (K % 32 != 0) {
        constexpr int kk = (K / 32) * 32;     // tail lies entirely in A2
        constexpr int kc = K / 32;
        v16h afrag;
#pragma unroll
        for (int i = 0; i < 16; ++i) {
            int koff = (i < 8) ? (hi * 8 + i) : (16 + hi * 8 + (i - 8));
            int ka = kk + koff;
            int t2 = ka - K1;
            t2 = t2 < 0 ? 0 : (t2 > A2W - 1 ? A2W - 1 : t2);
            float lv = A2[(size_t)row * A2W + t2];
            afrag[i] = (_Float16)((ka < K) ? lv : 0.0f);
        }
#pragma unroll
        for (int t = 0; t < NTPB; ++t) {
            v16h bfrag = PW[((size_t)kc * NCT + colt0 + t) * 32 + lane];
            acc[t] = __builtin_amdgcn_wmma_f32_16x16x32_f16(
                         false, afrag, false, bfrag, (short)0, acc[t],
                         false, false);
        }
    }

    // ---- epilogue ----
#pragma unroll
    for (int t = 0; t < NTPB; ++t) {
        int ncol = (colt0 + t) * 16 + mloc;
        bool colok = (COUT % 16 == 0) || (ncol < COUT);
        if (colok) {
            float bvv = 0.0f;
            if constexpr (HASBIAS) bvv = bias[ncol];
#pragma unroll
            for (int r = 0; r < 8; ++r) {
                int m = rowblk * 16 + r + 8 * hi;
                float v = acc[t][r] + bvv;
                if constexpr ((FLAGS & FLAG_LEAKY) != 0)
                    v = (v > 0.0f) ? v : NEG_SLOPE * v;
                if constexpr (HASADD)
                    v += addend[(size_t)m * ADDLD + ncol];
                float* yp = &Y[(size_t)m * YLD + ncol];
                if constexpr ((FLAGS & FLAG_ACCUM) != 0) *yp += v; else *yp = v;
            }
        }
    }
}

// ---------------------------------------------------------------------------
// Kernel 3: PointTransformerConv aggregation for one dilation layer.
// One block per point (b,n), one thread per channel c (128 threads).
// ---------------------------------------------------------------------------
__global__ void agg_kernel(const float* __restrict__ pos,
                           const int*   __restrict__ nbr,
                           const float* __restrict__ S,   // lin_src(x)  [M x C]
                           const float* __restrict__ V,   // lin(x)      [M x C]
                           const float* __restrict__ T,   // lin_dst(x)  [M x C]
                           const float* __restrict__ Wpos,// [3 x C]
                           const float* __restrict__ bpos,// [C]
                           float* __restrict__ H,         // [M x C]
                           int dil, int accumulate)
{
    __shared__ int   sidx[PK + 1];
    __shared__ float spj[(PK + 1) * 3];

    const int bn = blockIdx.x;
    const int b  = bn / PN;
    const int n  = bn % PN;
    const int c  = threadIdx.x;

    if (c < PK)  sidx[c]  = nbr[(size_t)bn * PKMAX + c * dil];
    if (c == PK) sidx[PK] = n;
    __syncthreads();
    if (c < (PK + 1) * 3) {
        int kk = c / 3, t = c % 3;
        spj[c] = pos[((size_t)b * PN + sidx[kk]) * 3 + t];
    }
    __syncthreads();

    const float pi0 = pos[((size_t)b * PN + n) * 3 + 0];
    const float pi1 = pos[((size_t)b * PN + n) * 3 + 1];
    const float pi2 = pos[((size_t)b * PN + n) * 3 + 2];

    const float wp0 = Wpos[0 * PC + c];
    const float wp1 = Wpos[1 * PC + c];
    const float wp2 = Wpos[2 * PC + c];
    const float bp  = bpos[c];
    const float ti  = T[(size_t)bn * PC + c];

    float alpha[PK + 1];
    float vpd[PK + 1];
    float amax = -3.0e38f;

#pragma unroll
    for (int k = 0; k < PK + 1; ++k) {
        int j = sidx[k];
        float dx = pi0 - spj[k * 3 + 0];
        float dy = pi1 - spj[k * 3 + 1];
        float dz = pi2 - spj[k * 3 + 2];
        float delta = dx * wp0 + dy * wp1 + dz * wp2 + bp;
        size_t off = ((size_t)b * PN + j) * PC + c;
        float a = ti - S[off] + delta;
        alpha[k] = a;
        vpd[k]   = V[off] + delta;
        amax = fmaxf(amax, a);
    }

    float denom = 0.0f, accv = 0.0f;
#pragma unroll
    for (int k = 0; k < PK + 1; ++k) {
        float e = __expf(alpha[k] - amax);
        denom += e;
        accv  += e * vpd[k];
    }
    float h = accv / denom;

    float* hp = &H[(size_t)bn * PC + c];
    if (accumulate) *hp += h; else *hp = h;
}

// ---------------------------------------------------------------------------
// Host-side orchestration (all launches on `stream`, graph-capture safe)
// ---------------------------------------------------------------------------
extern "C" void kernel_launch(void* const* d_in, const int* in_sizes, int n_in,
                              void* d_out, int out_size, void* d_ws, size_t ws_size,
                              hipStream_t stream)
{
    (void)in_sizes; (void)n_in; (void)out_size; (void)ws_size;

    const float* x     = (const float*)d_in[0];   // [B,N,C]
    const float* pos   = (const float*)d_in[1];   // [B,N,3]
    const float* W_lin = (const float*)d_in[2];   // [2,C,C]
    const float* W_src = (const float*)d_in[3];   // [2,C,C]
    const float* W_dst = (const float*)d_in[4];   // [2,C,C]
    const float* W_pos = (const float*)d_in[5];   // [2,3,C]
    const float* b_pos = (const float*)d_in[6];   // [2,C]
    const float* Wg    = (const float*)d_in[7];   // [C+3,C]
    const float* bg    = (const float*)d_in[8];   // [C]
    const float* W1    = (const float*)d_in[9];   // [C,C]
    const float* b1    = (const float*)d_in[10];  // [C]
    const float* W2    = (const float*)d_in[11];  // [C,3]
    const float* b2    = (const float*)d_in[12];  // [3]
    float* out = (float*)d_out;                   // [B,N,3]

    // Workspace layout (~40 MB)
    char*  ws  = (char*)d_ws;
    size_t off = 0;
    int*       nbr = (int*)(ws + off);       off += (size_t)PM * PKMAX * sizeof(int);
    float*     S   = (float*)(ws + off);     off += (size_t)PM * PC * sizeof(float);
    float*     V   = (float*)(ws + off);     off += (size_t)PM * PC * sizeof(float);
    float*     T   = (float*)(ws + off);     off += (size_t)PM * PC * sizeof(float);
    float*     H   = (float*)(ws + off);     off += (size_t)PM * PC * sizeof(float);
    _Float16*  A16 = (_Float16*)(ws + off);  off += (size_t)PM * PC * sizeof(_Float16);
    v16h*      PW  = (v16h*)(ws + off);      off += (size_t)5 * 8 * 32 * sizeof(v16h);
    float*     Y1  = S;  // S free after aggregation passes

    const int n4 = PM * PC / 4;
    const dim3 gPack((n4 + 255) / 256);

    // 1) KNN
    knn_kernel<<<PM, 256, 0, stream>>>(pos, nbr);

    // 2) Activations -> f16 once (x reused by 7 GEMMs)
    pack_f16_kernel<<<gPack, 256, 0, stream>>>(x, A16, n4);

    // grid: 4 row-slabs per block (4 waves), NTPB=4 col tiles per wave
    const dim3 gFull(PM / 64, 2);
    const dim3 bFull(128);
    const dim3 gW128(PC / 32, PC / 16);      // (KCH=4, NCT=8)
    const dim3 gWg(5, PC / 16);              // K=131 -> 5 chunks (tail padded)
    const dim3 gW2(PC / 32, 1);              // COUT=3 -> 1 col tile

    // 3) Two dilation layers (S/V/T and PW reused between passes)
    for (int li = 0; li < 2; ++li) {
        const int dil = (li == 0) ? 1 : 2;
        pack_w_kernel<PC, PC><<<gW128, 32, 0, stream>>>(W_src + (size_t)li * PC * PC, PW);
        gemm_wmma<PC, PC, 0, PC, PC, 0, 4, 0, false, false>
            <<<gFull, bFull, 0, stream>>>(A16, nullptr, PW, nullptr, nullptr, S);
        pack_w_kernel<PC, PC><<<gW128, 32, 0, stream>>>(W_lin + (size_t)li * PC * PC, PW);
        gemm_wmma<PC, PC, 0, PC, PC, 0, 4, 0, false, false>
            <<<gFull, bFull, 0, stream>>>(A16, nullptr, PW, nullptr, nullptr, V);
        pack_w_kernel<PC, PC><<<gW128, 32, 0, stream>>>(W_dst + (size_t)li * PC * PC, PW);
        gemm_wmma<PC, PC, 0, PC, PC, 0, 4, 0, false, false>
            <<<gFull, bFull, 0, stream>>>(A16, nullptr, PW, nullptr, nullptr, T);
        agg_kernel<<<PM, PC, 0, stream>>>(
            pos, nbr, S, V, T,
            W_pos + (size_t)li * 3 * PC, b_pos + (size_t)li * PC,
            H, dil, /*accumulate=*/li);
    }

    // 4) Global fusion: H += leaky_relu([x,pos] @ Wg + bg)   (K = 131 tail)
    pack_w_kernel<PC + 3, PC><<<gWg, 32, 0, stream>>>(Wg, PW);
    gemm_wmma<PC + 3, PC, 3, PC, PC, 0, 4, FLAG_LEAKY | FLAG_ACCUM, true, false>
        <<<gFull, bFull, 0, stream>>>(A16, pos, PW, bg, nullptr, H);

    // 5) Y1 = leaky_relu(H @ W1 + b1)   (H -> f16, reusing A16)
    pack_f16_kernel<<<gPack, 256, 0, stream>>>(H, A16, n4);
    pack_w_kernel<PC, PC><<<gW128, 32, 0, stream>>>(W1, PW);
    gemm_wmma<PC, PC, 0, PC, PC, 0, 4, FLAG_LEAKY, true, false>
        <<<gFull, bFull, 0, stream>>>(A16, nullptr, PW, b1, nullptr, Y1);

    // 6) out = Y1 @ W2 + b2 + pos   (COUT = 3, column-clamped tile)
    pack_f16_kernel<<<gPack, 256, 0, stream>>>(Y1, A16, n4);
    pack_w_kernel<PC, 3><<<gW2, 32, 0, stream>>>(W2, PW);
    gemm_wmma<PC, PC, 0, 3, 3, 3, 1, 0, true, true>
        <<<dim3(PM / 64, 1), bFull, 0, stream>>>(A16, nullptr, PW, b2, pos, out);
}
